// SAKT_51118700757522
// MI455X (gfx1250) — compile-verified, matrix operand order checked by reading, not run on hardware
//
#include <hip/hip_runtime.h>
#include <hip/hip_fp16.h>
#include <hip/hip_bf16.h>

#define B_  16
#define L_  512
#define D_  512
#define H_  8
#define NL_ 4
#define FF_ 2048
#define HD_ 64
#define PLD 520   // padded prob-row stride (halves) to avoid LDS bank conflicts

typedef _Float16 v8h  __attribute__((ext_vector_type(8)));
typedef _Float16 v16h __attribute__((ext_vector_type(16)));
typedef float    v8f  __attribute__((ext_vector_type(8)));

// ---- WMMA helpers -----------------------------------------------------------
// 16-bit A/B fragment: lane = lo + 16*hi ; element e -> K = (e&7) + 8*hi + 16*(e>>3)
// => two contiguous 8-half (16B) loads at (base + kk + 8*hi) and (+16).
__device__ __forceinline__ v16h load_frag(const _Float16* p) {
    v8h a = *(const v8h*)(p);
    v8h b = *(const v8h*)(p + 16);
    v16h f;
#pragma unroll
    for (int i = 0; i < 8; ++i) { f[i] = a[i]; f[i + 8] = b[i]; }
    return f;
}

__device__ __forceinline__ v8f wmma_f16(v16h a, v16h b, v8f c) {
    return __builtin_amdgcn_wmma_f32_16x16x32_f16(false, a, false, b, (short)0, c,
                                                  false, false);
}

__device__ __forceinline__ float wave_sum(float v) {
#pragma unroll
    for (int o = 16; o; o >>= 1) v += __shfl_xor(v, o, 32);
    return v;
}
__device__ __forceinline__ float wave_max(float v) {
#pragma unroll
    for (int o = 16; o; o >>= 1) v = fmaxf(v, __shfl_xor(v, o, 32));
    return v;
}

// ---- GEMM: C[M,N] = A[M,K] (f16) @ Bt[N,K]^T (f16) + bias, opt ReLU ---------
// Block = 256 threads = 8 waves as 4(M) x 2(N); wave tile 32x64; block 128x128.
// Ping-pong unroll-by-2 over K (K % 64 == 0): WMMAs on one fragment set while
// loads fill the disjoint other set -> no register copies, loads overlap math.
#define GF_BIAS 1
#define GF_RELU 2
#define GF_F32  4
#define GF_F16  8

__device__ __forceinline__ void load_ab(const _Float16* arow0, const _Float16* arow1,
                                        const _Float16* brow, size_t bstr, int kk,
                                        v16h a[2], v16h b[4])
{
    a[0] = load_frag(arow0 + kk);
    a[1] = load_frag(arow1 + kk);
#pragma unroll
    for (int nt = 0; nt < 4; ++nt) b[nt] = load_frag(brow + bstr * nt + kk);
}

__device__ __forceinline__ void mma8(v8f acc[2][4], const v16h a[2], const v16h b[4])
{
#pragma unroll
    for (int mt = 0; mt < 2; ++mt)
#pragma unroll
        for (int nt = 0; nt < 4; ++nt)
            acc[mt][nt] = wmma_f16(a[mt], b[nt], acc[mt][nt]);
}

__global__ __launch_bounds__(256) void gemm_kernel(
    const _Float16* __restrict__ A, const _Float16* __restrict__ Bt,
    const float* __restrict__ bias, float* __restrict__ Cf,
    _Float16* __restrict__ Ch, int M, int N, int K, int flags)
{
    const int lane = threadIdx.x & 31;
    const int wave = threadIdx.x >> 5;
    const int lo = lane & 15, hi = lane >> 4;
    const int wm = wave & 3, wn = wave >> 2;
    const int m0 = blockIdx.x * 128 + wm * 32;
    const int n0 = blockIdx.y * 128 + wn * 64;
    if (m0 >= M) return;

    v8f acc[2][4] = {};
    const _Float16* arow0 = A + (size_t)(m0 + lo) * K + 8 * hi;
    const _Float16* arow1 = arow0 + (size_t)16 * K;
    const _Float16* brow = Bt + (size_t)(n0 + lo) * K + 8 * hi;
    const size_t bstr = (size_t)16 * K;

    v16h a0[2], b0[4], a1[2], b1[4];
    load_ab(arow0, arow1, brow, bstr, 0, a0, b0);
    for (int kk = 0; kk < K; kk += 64) {
        load_ab(arow0, arow1, brow, bstr, kk + 32, a1, b1);
        mma8(acc, a0, b0);
        if (kk + 64 < K) load_ab(arow0, arow1, brow, bstr, kk + 64, a0, b0);
        mma8(acc, a1, b1);
    }

#pragma unroll
    for (int nt = 0; nt < 4; ++nt) {
        const int n = n0 + nt * 16 + lo;
        const float bv = (flags & GF_BIAS) ? bias[n] : 0.f;
#pragma unroll
        for (int mt = 0; mt < 2; ++mt)
#pragma unroll
            for (int r = 0; r < 8; ++r) {
                const int m = m0 + mt * 16 + r + 8 * hi;
                float v = acc[mt][nt][r] + bv;
                if (flags & GF_RELU) v = fmaxf(v, 0.f);
                const size_t off = (size_t)m * N + n;
                if (flags & GF_F32) Cf[off] = v;
                if (flags & GF_F16) Ch[off] = (_Float16)v;
            }
    }
}

// ---- Fused causal attention: one wave per (qtile, head, batch) --------------
// Q,K: [B,L,H*HD] f16 ; Vt: [B,H,HD,L] f16 ; ctx: [B,L,H*HD] f16
__global__ __launch_bounds__(32) void attn_kernel(
    const _Float16* __restrict__ Q, const _Float16* __restrict__ Km,
    const _Float16* __restrict__ Vt, _Float16* __restrict__ ctx,
    int L, int H, int HD, float inv_scale)
{
    __shared__ float Sf[16 * 512];
    __shared__ __align__(16) _Float16 P[16 * PLD];

    const int lane = threadIdx.x;
    const int lo = lane & 15, hi = lane >> 4;
    const int b = blockIdx.z, h = blockIdx.y, qt = blockIdx.x;
    const int m0 = qt * 16;
    const int Dm = H * HD;

    const _Float16* qbase = Q + ((size_t)b * L + m0) * Dm + h * HD;
    const _Float16* kbase = Km + (size_t)b * L * Dm + h * HD;

    // Phase 1: scores S = Q K^T * inv_scale, causal-masked, only needed tiles
    const int ntiles = qt + 1;
    const _Float16* qrow = qbase + (size_t)lo * Dm + 8 * hi;
    for (int nt = 0; nt < ntiles; ++nt) {
        v8f acc = {};
        const _Float16* krow = kbase + (size_t)(nt * 16 + lo) * Dm + 8 * hi;
        for (int kk = 0; kk < HD; kk += 32)
            acc = wmma_f16(load_frag(qrow + kk), load_frag(krow + kk), acc);
        const int n = nt * 16 + lo;
#pragma unroll
        for (int r = 0; r < 8; ++r) {
            const int qi = m0 + r + 8 * hi;
            Sf[(r + 8 * hi) * 512 + n] = (n <= qi) ? acc[r] * inv_scale : -1e30f;
        }
    }

    // Phase 2: row softmax (f32) -> probs (f16, zero-padded to 32-aligned K)
    const int ncols = 16 * ntiles;
    const int kend = (ncols + 31) & ~31;
    for (int r = 0; r < 16; ++r) {
        float mx = -1e30f;
        for (int c = lane; c < ncols; c += 32) mx = fmaxf(mx, Sf[r * 512 + c]);
        mx = wave_max(mx);
        float sum = 0.f;
        for (int c = lane; c < ncols; c += 32) {
            float e = __expf(Sf[r * 512 + c] - mx);
            Sf[r * 512 + c] = e;
            sum += e;
        }
        sum = wave_sum(sum);
        const float inv = 1.f / sum;
        for (int c = lane; c < ncols; c += 32)
            P[r * PLD + c] = (_Float16)(Sf[r * 512 + c] * inv);
        for (int c = ncols + lane; c < kend; c += 32)
            P[r * PLD + c] = (_Float16)0.f;
    }

    // Phase 3: ctx = P @ V, output in chunks of 64 columns (VGPR-bounded)
    const _Float16* vbase = Vt + ((size_t)b * H + h) * (size_t)HD * L;
    _Float16* cbase = ctx + ((size_t)b * L + m0) * Dm + h * HD;
    const _Float16* prow = &P[lo * PLD + 8 * hi];
    for (int nc = 0; nc < HD; nc += 64) {
        v8f acc[4] = {};
        for (int kk = 0; kk < kend; kk += 32) {
            v16h af = load_frag(prow + kk);
#pragma unroll
            for (int nt = 0; nt < 4; ++nt) {
                const _Float16* vrow =
                    vbase + (size_t)(nc + nt * 16 + lo) * L + 8 * hi + kk;
                acc[nt] = wmma_f16(af, load_frag(vrow), acc[nt]);
            }
        }
#pragma unroll
        for (int nt = 0; nt < 4; ++nt)
#pragma unroll
            for (int r = 0; r < 8; ++r)
                cbase[(size_t)(r + 8 * hi) * Dm + nc + nt * 16 + lo] =
                    (_Float16)acc[nt][r];
    }
}

// ---- residual add + LayerNorm (wave per row of 512) -------------------------
__global__ __launch_bounds__(256) void add_ln_kernel(
    const float* __restrict__ X, const float* __restrict__ Y,
    const float* __restrict__ g, const float* __restrict__ be,
    float* __restrict__ Xo, _Float16* __restrict__ Xh, int rows)
{
    const int wave = threadIdx.x >> 5, lane = threadIdx.x & 31;
    const int row = blockIdx.x * 8 + wave;
    if (row >= rows) return;
    const float* xr = X + (size_t)row * D_;
    const float* yr = Y + (size_t)row * D_;
    float v[16];
    float s = 0.f;
#pragma unroll
    for (int i = 0; i < 16; ++i) {
        const int c = lane + i * 32;
        v[i] = xr[c] + yr[c];
        s += v[i];
    }
    s = wave_sum(s);
    const float mean = s * (1.f / (float)D_);
    float q = 0.f;
#pragma unroll
    for (int i = 0; i < 16; ++i) { const float d = v[i] - mean; q += d * d; }
    q = wave_sum(q);
    const float rstd = rsqrtf(q * (1.f / (float)D_) + 1e-5f);
    float* xo = Xo + (size_t)row * D_;
    _Float16* xh = Xh + (size_t)row * D_;
#pragma unroll
    for (int i = 0; i < 16; ++i) {
        const int c = lane + i * 32;
        const float o = (v[i] - mean) * rstd * g[c] + be[c];
        xo[c] = o;
        xh[c] = (_Float16)o;
    }
}

// ---- embedding: q_table[q] + r_table[r] + p_table[l], write f32 + f16 -------
__global__ void embed_kernel(const int* __restrict__ q, const int* __restrict__ r,
                             const float* __restrict__ qt, const float* __restrict__ rt,
                             const float* __restrict__ pt,
                             float* __restrict__ x, _Float16* __restrict__ xh)
{
    const int idx = blockIdx.x * blockDim.x + threadIdx.x;  // over B*L*D (exact)
    const int c = idx % D_;
    const int row = idx / D_;
    const int l = row % L_;
    const float v = qt[(size_t)q[row] * D_ + c] + rt[(size_t)r[row] * D_ + c] +
                    pt[(size_t)l * D_ + c];
    x[idx] = v;
    xh[idx] = (_Float16)v;
}

__global__ void gatherq_kernel(const int* __restrict__ q,
                               const float* __restrict__ qt,
                               _Float16* __restrict__ out)
{
    const int idx = blockIdx.x * blockDim.x + threadIdx.x;
    const int c = idx % D_;
    const int row = idx / D_;
    out[idx] = (_Float16)qt[(size_t)q[row] * D_ + c];
}

// ---- weight transpose+convert: W[K,N] f32 -> Wt[N,K] f16 --------------------
__global__ void wt_kernel(const float* __restrict__ W, _Float16* __restrict__ Wt,
                          int K, int N)
{
    const int idx = blockIdx.x * blockDim.x + threadIdx.x;
    if (idx >= N * K) return;
    const int k = idx % K;
    const int n = idx / K;
    Wt[idx] = (_Float16)W[(size_t)k * N + n];
}

// ---- V transpose: [B,L,H,HD] f16 -> [B,H,HD,L] f16 --------------------------
__global__ void vt_kernel(const _Float16* __restrict__ Vh,
                          _Float16* __restrict__ Vt, int H, int HD)
{
    const int idx = blockIdx.x * blockDim.x + threadIdx.x;  // over B*L*D (exact)
    const int l = idx % L_;
    int t = idx / L_;
    const int d = t % HD;
    t /= HD;
    const int h = t % H;
    const int b = t / H;
    Vt[idx] = Vh[(((size_t)b * L_ + l) * H + h) * HD + d];
}

// ---- final head: sigmoid(fc1 @ Wf2 + bf2), wave per row ---------------------
__global__ __launch_bounds__(256) void fc2_kernel(
    const _Float16* __restrict__ fc1, const float* __restrict__ w,
    const float* __restrict__ bias, float* __restrict__ out, int rows)
{
    const int wave = threadIdx.x >> 5, lane = threadIdx.x & 31;
    const int row = blockIdx.x * 8 + wave;
    if (row >= rows) return;
    float s = 0.f;
#pragma unroll
    for (int i = 0; i < 8; ++i) {
        const int c = lane + i * 32;
        s += (float)fc1[(size_t)row * 256 + c] * w[c];
    }
    s = wave_sum(s);
    if (lane == 0) out[row] = 1.f / (1.f + __expf(-(s + bias[0])));
}

// ---- host driver ------------------------------------------------------------
extern "C" void kernel_launch(void* const* d_in, const int* in_sizes, int n_in,
                              void* d_out, int out_size, void* d_ws, size_t ws_size,
                              hipStream_t stream)
{
    (void)in_sizes; (void)n_in; (void)out_size;

    const int*   questions = (const int*)d_in[0];
    const int*   responses = (const int*)d_in[1];
    const float* q_table   = (const float*)d_in[2];
    const float* r_table   = (const float*)d_in[3];
    const float* p_table   = (const float*)d_in[4];
    const float* Wq = (const float*)d_in[5];   const float* bq = (const float*)d_in[6];
    const float* Wk = (const float*)d_in[7];   const float* bk = (const float*)d_in[8];
    const float* Wv = (const float*)d_in[9];   const float* bv = (const float*)d_in[10];
    const float* Wo = (const float*)d_in[11];  const float* bo = (const float*)d_in[12];
    const float* W1 = (const float*)d_in[13];  const float* b1 = (const float*)d_in[14];
    const float* W2 = (const float*)d_in[15];  const float* b2 = (const float*)d_in[16];
    const float* g1 = (const float*)d_in[17];  const float* be1 = (const float*)d_in[18];
    const float* g2 = (const float*)d_in[19];  const float* be2 = (const float*)d_in[20];
    const float* Wpq = (const float*)d_in[21]; const float* bpq = (const float*)d_in[22];
    const float* Wpk = (const float*)d_in[23]; const float* bpk = (const float*)d_in[24];
    const float* Wpv = (const float*)d_in[25]; const float* bpv = (const float*)d_in[26];
    const float* Wf1 = (const float*)d_in[27]; const float* bf1 = (const float*)d_in[28];
    const float* Wf2 = (const float*)d_in[29]; const float* bf2 = (const float*)d_in[30];
    float* out = (float*)d_out;

    const int Mrows = B_ * L_;
    char* ws = (char*)d_ws;
    size_t off = 0;
    auto carve = [&](size_t bytes) -> char* {
        char* p = ws + off;
        off += (bytes + 255) & ~(size_t)255;
        return p;
    };
    float*    xf   = (float*)carve((size_t)Mrows * D_ * 4);
    _Float16* xh   = (_Float16*)carve((size_t)Mrows * D_ * 2);
    float*    yf   = (float*)carve((size_t)Mrows * D_ * 4);
    _Float16* qh   = (_Float16*)carve((size_t)Mrows * D_ * 2);  // hbuf aliases
    _Float16* kh   = (_Float16*)carve((size_t)Mrows * D_ * 2);  // qh..vt (32MB)
    _Float16* vh   = (_Float16*)carve((size_t)Mrows * D_ * 2);
    _Float16* vt   = (_Float16*)carve((size_t)Mrows * D_ * 2);
    _Float16* ctxh = (_Float16*)carve((size_t)Mrows * D_ * 2);
    _Float16* WqT  = (_Float16*)carve((size_t)NL_ * D_ * D_ * 2);
    _Float16* WkT  = (_Float16*)carve((size_t)NL_ * D_ * D_ * 2);
    _Float16* WvT  = (_Float16*)carve((size_t)NL_ * D_ * D_ * 2);
    _Float16* WoT  = (_Float16*)carve((size_t)NL_ * D_ * D_ * 2);
    _Float16* W1T  = (_Float16*)carve((size_t)NL_ * D_ * FF_ * 2);
    _Float16* W2T  = (_Float16*)carve((size_t)NL_ * D_ * FF_ * 2);
    _Float16* WpqT = (_Float16*)carve((size_t)D_ * D_ * 2);
    _Float16* WpkT = (_Float16*)carve((size_t)D_ * D_ * 2);
    _Float16* WpvT = (_Float16*)carve((size_t)D_ * D_ * 2);
    _Float16* Wf1T = (_Float16*)carve((size_t)256 * D_ * 2);
    if (off > ws_size) return;  // insufficient workspace
    _Float16* hbuf = qh;        // [Mrows, FF_] f16, aliases qh..vt region

    auto wt = [&](const float* W, _Float16* Wt, int K, int N) {
        const int total = K * N;
        wt_kernel<<<(total + 255) / 256, 256, 0, stream>>>(W, Wt, K, N);
    };
    auto gemm = [&](const _Float16* A, const _Float16* Bt, const float* bias,
                    float* Cf, _Float16* Ch, int M, int N, int K, int flags) {
        dim3 grid(M / 128, N / 128);
        gemm_kernel<<<grid, 256, 0, stream>>>(A, Bt, bias, Cf, Ch, M, N, K, flags);
    };

    // ---- weight prep (transpose + f16 convert), every call (deterministic)
    for (int i = 0; i < NL_; ++i) {
        const size_t wo = (size_t)i * D_ * D_;
        wt(Wq + wo, WqT + wo, D_, D_);
        wt(Wk + wo, WkT + wo, D_, D_);
        wt(Wv + wo, WvT + wo, D_, D_);
        wt(Wo + wo, WoT + wo, D_, D_);
        wt(W1 + (size_t)i * D_ * FF_, W1T + (size_t)i * D_ * FF_, D_, FF_);
        wt(W2 + (size_t)i * FF_ * D_, W2T + (size_t)i * FF_ * D_, FF_, D_);
    }
    wt(Wpq, WpqT, D_, D_);
    wt(Wpk, WpkT, D_, D_);
    wt(Wpv, WpvT, D_, D_);
    wt(Wf1, Wf1T, D_, 256);

    // ---- embedding
    embed_kernel<<<(Mrows * D_) / 256, 256, 0, stream>>>(
        questions, responses, q_table, r_table, p_table, xf, xh);

    // ---- transformer layers
    for (int i = 0; i < NL_; ++i) {
        const size_t wo = (size_t)i * D_ * D_;
        gemm(xh, WqT + wo, bq + i * D_, nullptr, qh, Mrows, D_, D_, GF_BIAS | GF_F16);
        gemm(xh, WkT + wo, bk + i * D_, nullptr, kh, Mrows, D_, D_, GF_BIAS | GF_F16);
        gemm(xh, WvT + wo, bv + i * D_, nullptr, vh, Mrows, D_, D_, GF_BIAS | GF_F16);
        vt_kernel<<<(Mrows * D_) / 256, 256, 0, stream>>>(vh, vt, H_, HD_);
        attn_kernel<<<dim3(L_ / 16, H_, B_), 32, 0, stream>>>(
            qh, kh, vt, ctxh, L_, H_, HD_, 0.125f);  // 1/sqrt(64)
        gemm(ctxh, WoT + wo, bo + i * D_, yf, nullptr, Mrows, D_, D_, GF_BIAS | GF_F32);
        add_ln_kernel<<<Mrows / 8, 256, 0, stream>>>(xf, yf, g1 + i * D_, be1 + i * D_,
                                                     xf, xh, Mrows);
        gemm(xh, W1T + (size_t)i * D_ * FF_, b1 + i * FF_, nullptr, hbuf,
             Mrows, FF_, D_, GF_BIAS | GF_RELU | GF_F16);
        gemm(hbuf, W2T + (size_t)i * D_ * FF_, b2 + i * D_, yf, nullptr,
             Mrows, D_, FF_, GF_BIAS | GF_F32);
        add_ln_kernel<<<Mrows / 8, 256, 0, stream>>>(xf, yf, g2 + i * D_, be2 + i * D_,
                                                     xf, xh, Mrows);
    }

    // ---- prediction head: single-head attention over full D
    _Float16* qemb = (_Float16*)yf;  // reuse (yf free now)
    gatherq_kernel<<<(Mrows * D_) / 256, 256, 0, stream>>>(questions, q_table, qemb);
    gemm(qemb, WpqT, bpq, nullptr, qh, Mrows, D_, D_, GF_BIAS | GF_F16);
    gemm(xh, WpkT, bpk, nullptr, kh, Mrows, D_, D_, GF_BIAS | GF_F16);
    gemm(xh, WpvT, bpv, nullptr, vh, Mrows, D_, D_, GF_BIAS | GF_F16);
    vt_kernel<<<(Mrows * D_) / 256, 256, 0, stream>>>(vh, vt, 1, D_);
    attn_kernel<<<dim3(L_ / 16, 1, B_), 32, 0, stream>>>(
        qh, kh, vt, ctxh, L_, 1, D_, 0.04419417382415922f);  // 1/sqrt(512)

    _Float16* fc1 = (_Float16*)yf;  // [Mrows,256] f16, reuse again
    gemm(ctxh, Wf1T, bf1, nullptr, fc1, Mrows, 256, D_, GF_BIAS | GF_RELU | GF_F16);
    fc2_kernel<<<Mrows / 8, 256, 0, stream>>>(fc1, Wf2, bf2, out, Mrows);
}
